// CentroidLossExponential_58729382806235
// MI455X (gfx1250) — compile-verified, matrix operand order checked by reading, not run on hardware
//
#include <hip/hip_runtime.h>
#include <hip/hip_bf16.h>
#include <math.h>

// Problem constants (match reference)
#define B_DIM 32768
#define D_DIM 256
#define C_DIM 2048
#define EPS_F 1e-12f
#define NCHUNK (C_DIM / 16)

typedef __attribute__((ext_vector_type(16))) _Float16 v16h;
typedef __attribute__((ext_vector_type(8)))  _Float16 v8h;
typedef __attribute__((ext_vector_type(8)))  float    v8f;
typedef __attribute__((ext_vector_type(4)))  int      v4i;
typedef __attribute__((ext_vector_type(4)))  unsigned int u32x4;
typedef __attribute__((ext_vector_type(8)))  int      i32x8;
typedef __attribute__((ext_vector_type(4)))  int      i32x4;

// Pointer-to-v4i in global (AS1) / LDS (AS3) address spaces, as required by
// the gfx1250 async-load builtins (param type confirmed by compiler diag).
typedef __attribute__((address_space(1))) v4i* gv4i_ptr;
typedef __attribute__((address_space(3))) v4i* lv4i_ptr;

// ---------------------------------------------------------------------------
// CDNA5 data movers, in preference order:
//   1) Tensor Data Mover (one DMA instruction per 8 KB tile, TENSORcnt)
//   2) per-lane async global->LDS (ASYNCcnt)
//   3) sync copy through VGPRs
// ---------------------------------------------------------------------------
#if defined(__has_builtin)
#  if __has_builtin(__builtin_amdgcn_tensor_load_to_lds)
#    define HAVE_TDM 1
#  endif
#  if __has_builtin(__builtin_amdgcn_global_load_async_to_lds_b128)
#    define HAVE_ASYNC_LDS 1
#  endif
#endif
#ifndef HAVE_TDM
#  define HAVE_TDM 0
#endif
#ifndef HAVE_ASYNC_LDS
#  define HAVE_ASYNC_LDS 0
#endif

#if HAVE_ASYNC_LDS
#  if __has_builtin(__builtin_amdgcn_s_wait_asynccnt)
#    define WAIT_ASYNC(n) __builtin_amdgcn_s_wait_asynccnt(n)
#  else
#    define WAIT_ASYNC(n) asm volatile("s_wait_asynccnt %0" :: "i"(n) : "memory")
#  endif
#else
#  define WAIT_ASYNC(n) do { } while (0)
#endif

// ---------------------------------------------------------------------------
// Workspace layout (bytes)
// ---------------------------------------------------------------------------
#define OFF_FN16   (0)
#define OFF_CENT16 (OFF_FN16   + (size_t)B_DIM * D_DIM * 2)
#define OFF_SUMS   (OFF_CENT16 + (size_t)C_DIM * D_DIM * 2)
#define OFF_COUNTS (OFF_SUMS   + (size_t)C_DIM * D_DIM * 4)
#define OFF_CN2    (OFF_COUNTS + (size_t)C_DIM * 4)
#define OFF_DN2    (OFF_CN2    + (size_t)C_DIM * 4)
#define OFF_ACCUM  (OFF_DN2    + (size_t)B_DIM * 4)

// ---------------------------------------------------------------------------
// Kernel 0: zero/initialize workspace accumulators
// ---------------------------------------------------------------------------
__global__ __launch_bounds__(256) void cl_init_kernel(
    float* __restrict__ sums, float* __restrict__ counts,
    unsigned int* __restrict__ dn2min, float* __restrict__ accum)
{
    const int idx = blockIdx.x * 256 + threadIdx.x;
    if (idx < C_DIM * D_DIM) sums[idx] = 0.0f;
    if (idx < C_DIM)         counts[idx] = 0.0f;
    if (idx < B_DIM)         dn2min[idx] = 0x7F800000u;   // +inf bits
    if (idx < 2)             accum[idx] = 0.0f;
}

// ---------------------------------------------------------------------------
// Kernel 1: L2-normalize rows; write f16 A-matrix (= -2*f); atomic segment sums
// ---------------------------------------------------------------------------
__global__ __launch_bounds__(256) void cl_norm_scatter_kernel(
    const float* __restrict__ feat, const int* __restrict__ labels,
    _Float16* __restrict__ fn16, float* __restrict__ sums,
    float* __restrict__ counts)
{
    __shared__ float red[256];
    const int row = blockIdx.x;
    const int i   = threadIdx.x;
    const float x = feat[(size_t)row * D_DIM + i];
    red[i] = x * x;
    __syncthreads();
    #pragma unroll
    for (int s = 128; s > 0; s >>= 1) {
        if (i < s) red[i] += red[i + s];
        __syncthreads();
    }
    const float nrm = sqrtf(red[0]);
    const float f   = x / fmaxf(nrm, EPS_F);
    fn16[(size_t)row * D_DIM + i] = (_Float16)(-2.0f * f);
    const int lab = labels[row];
    atomicAdd(&sums[(size_t)lab * D_DIM + i], f);
    if (i == 0) atomicAdd(&counts[lab], 1.0f);
}

// ---------------------------------------------------------------------------
// Kernel 2: centroids (f16) and cn2 = ||centroid_f16||^2. One block per class.
// ---------------------------------------------------------------------------
__global__ __launch_bounds__(256) void cl_centroid_kernel(
    const float* __restrict__ sums, const float* __restrict__ counts,
    _Float16* __restrict__ cent16, float* __restrict__ cn2)
{
    __shared__ float red[256];
    const int c = blockIdx.x;
    const int i = threadIdx.x;
    const float cnt = counts[c];
    const float cv  = sums[(size_t)c * D_DIM + i] / fmaxf(cnt, 1.0f);
    const _Float16 ch = (_Float16)cv;
    cent16[(size_t)c * D_DIM + i] = ch;
    const float cv16 = (float)ch;           // rounded value keeps d2 >= 0
    red[i] = cv16 * cv16;
    __syncthreads();
    #pragma unroll
    for (int s = 128; s > 0; s >>= 1) {
        if (i < s) red[i] += red[i + s];
        __syncthreads();
    }
    if (i == 0) cn2[c] = red[0];
}

// ---------------------------------------------------------------------------
// Tile staging: one 16-class x 256-half centroid tile (8 KB, contiguous).
// ---------------------------------------------------------------------------
#if HAVE_TDM
// Describe the 8 KB tile as a 1-D tensor (data_size=2B, dim0=tile0=4096,
// all higher dims = 1) and fire one TENSOR_LOAD_TO_LDS. Issued by wave 0 only.
__device__ __forceinline__ void cl_stage_tile_tdm(
    const _Float16* srcTile, _Float16* dstLds)
{
    const unsigned long long ga = (unsigned long long)(size_t)srcTile;
    const unsigned la = (unsigned)(size_t)(lv4i_ptr)(void*)dstLds;
    // D# group 0: count=1 | lds_addr | global_addr[56:0] | type=2
    u32x4 g0 = { 1u, la, (unsigned)ga,
                 (unsigned)((ga >> 32) & 0x01FFFFFFu) | (2u << 30) };
    // D# group 1: data_size=2B; tensor_dim0=4096; tensor_dim1=1; tile_dim0=4096
    i32x8 g1 = { (int)(1u << 16),        // [17:16] data_size = 1 (2 bytes)
                 (int)0x10000000,        // tensor_dim0[15:0]=4096 << 16
                 (int)0x00010000,        // tensor_dim0[31:16]=0 | tensor_dim1 lo=1
                 (int)0x10000000,        // tensor_dim1 hi=0 | tile_dim0=4096
                 0,                      // tile_dim1=0 (unused), tile_dim2=0
                 4096,                   // tensor_dim0_stride lo
                 (int)0x10000000,        // stride0 hi=0 | tensor_dim1_stride lo=4096
                 0 };                    // tensor_dim1_stride hi
    // D# group 2: tensor_dim2=1, tensor_dim3=1, stride2=4096, tile_dim3=0
    i32x4 g2 = { 1, 1, 4096, 0 };
    // D# group 3: tensor_dim3_stride=4096, tensor_dim4=1, tile_dim4=0
    i32x4 g3 = { 4096, (int)(1u << 16), 0, 0 };
#if __clang_major__ >= 23
    // amdgpu-toolchain (clang-23+) form: extra int32x8 operand, zero-filled
    i32x8 g4 = { 0, 0, 0, 0, 0, 0, 0, 0 };
    __builtin_amdgcn_tensor_load_to_lds(g0, g1, g2, g3, g4, 0);
#else
    // ROCm 7.2 (clang-22) form: 5 args
    __builtin_amdgcn_tensor_load_to_lds(g0, g1, g2, g3, 0);
#endif
}
#endif

// Per-lane path: each of the 256 threads moves 32 bytes = two b128 transfers.
__device__ __forceinline__ void cl_stage_tile(
    const _Float16* __restrict__ srcTile, _Float16* dstLds, int tid)
{
    const int cls = tid >> 4;            // 0..15
    const int ko  = (tid & 15) * 16;     // 0..240 halfs
    const _Float16* src = srcTile + (size_t)cls * D_DIM + ko;
    _Float16*       dst = dstLds + cls * D_DIM + ko;
#if HAVE_ASYNC_LDS
    __builtin_amdgcn_global_load_async_to_lds_b128(
        (gv4i_ptr)(void*)src,       (lv4i_ptr)(void*)dst,       0, 0);
    __builtin_amdgcn_global_load_async_to_lds_b128(
        (gv4i_ptr)(void*)(src + 8), (lv4i_ptr)(void*)(dst + 8), 0, 0);
#else
    v8h* d = (v8h*)dst;
    d[0] = ((const v8h*)src)[0];
    d[1] = ((const v8h*)src)[1];
#endif
}

__device__ __forceinline__ void cl_stage_issue(
    const _Float16* srcTile, _Float16* dstLds, int tid)
{
#if HAVE_TDM
    if (tid < 32) cl_stage_tile_tdm(srcTile, dstLds);   // wave 0 drives the DMA
#else
    cl_stage_tile(srcTile, dstLds, tid);
#endif
}

// One newer stage may remain outstanding (double buffer in flight).
__device__ __forceinline__ void cl_stage_wait_prev(int tid)
{
#if HAVE_TDM
    if (tid < 32) __builtin_amdgcn_s_wait_tensorcnt(1);
#elif HAVE_ASYNC_LDS
    WAIT_ASYNC(2);
#endif
}

__device__ __forceinline__ void cl_stage_wait_all(int tid)
{
#if HAVE_TDM
    if (tid < 32) __builtin_amdgcn_s_wait_tensorcnt(0);
#elif HAVE_ASYNC_LDS
    WAIT_ASYNC(0);
#endif
}

// ---------------------------------------------------------------------------
// Kernel 3: WMMA GEMM + masked row-min, double-buffered LDS B tiles.
// Block = 256 threads = 8 waves; covers 128 sample rows (16 per wave).
// d2 = (1 + cn2[col]) + (-2f) . c   via v_wmma_f32_16x16x32_f16 (K = 256).
// ---------------------------------------------------------------------------
__global__ __launch_bounds__(256) void cl_gemm_min_kernel(
    const _Float16* __restrict__ fn16,    // [B,D]  (= -2 * f)
    const _Float16* __restrict__ cent16,  // [C,D]
    const float* __restrict__ cn2,        // [C]
    const float* __restrict__ counts,     // [C]
    const int* __restrict__ labels,       // [B]
    unsigned int* __restrict__ dn2min)    // [B] monotone bits, init +inf
{
    __shared__ __align__(16) _Float16 ldsB[2][16 * D_DIM];   // 2 x 8 KB

    const int tid     = threadIdx.x;
    const int lane    = tid & 31;
    const int wave    = tid >> 5;
    const int rowW    = blockIdx.x * 128 + wave * 16;
    const int l16     = lane & 15;
    const int halfSel = (lane >> 4) & 1;
    const int hk      = halfSel * 8;

    // --- preload the full-K A fragments for this wave's 16 rows -------------
    v16h afrag[8];
    {
        const _Float16* ap = fn16 + (size_t)(rowW + l16) * D_DIM;
        #pragma unroll
        for (int kc = 0; kc < 8; ++kc) {
            const int kb = kc * 32 + hk;
            v8h lo = *(const v8h*)(ap + kb);
            v8h hi = *(const v8h*)(ap + kb + 16);
            afrag[kc] = __builtin_shufflevector(
                lo, hi, 0,1,2,3,4,5,6,7,8,9,10,11,12,13,14,15);
        }
    }

    int mylab[8];
    #pragma unroll
    for (int j = 0; j < 8; ++j)
        mylab[j] = labels[rowW + j + halfSel * 8];

    float minv[8];
    #pragma unroll
    for (int j = 0; j < 8; ++j) minv[j] = __builtin_inff();

    // prologue: stage chunk 0 into buffer 0
    cl_stage_issue(cent16, &ldsB[0][0], tid);

    for (int i = 0; i < NCHUNK; ++i) {
        const int nb = i * 16;

        // stage next chunk into the other buffer while we compute this one
        if (i + 1 < NCHUNK) {
            cl_stage_issue(cent16 + (size_t)(nb + 16) * D_DIM,
                           &ldsB[(i + 1) & 1][0], tid);
            cl_stage_wait_prev(tid);   // previous stage complete, newest in flight
        } else {
            cl_stage_wait_all(tid);
        }
        __syncthreads();      // publish staged tile across all 8 waves

        const _Float16* bufB = &ldsB[i & 1][0];
        const int   col   = nb + l16;
        const float bias  = 1.0f + cn2[col];
        const bool  empty = (counts[col] <= 0.0f);

        v8f acc;
        #pragma unroll
        for (int j = 0; j < 8; ++j) acc[j] = bias;

        #pragma unroll
        for (int kc = 0; kc < 8; ++kc) {
            const int kb = kc * 32 + hk;
            const _Float16* bp = bufB + l16 * D_DIM + kb;
            v8h lo = *(const v8h*)(bp);
            v8h hi = *(const v8h*)(bp + 16);
            v16h bfrag = __builtin_shufflevector(
                lo, hi, 0,1,2,3,4,5,6,7,8,9,10,11,12,13,14,15);
            acc = __builtin_amdgcn_wmma_f32_16x16x32_f16(
                false, afrag[kc], false, bfrag,
                (short)0, acc, false, false);
        }

        #pragma unroll
        for (int j = 0; j < 8; ++j) {
            const bool excl = empty || (mylab[j] == col);
            const float d2  = excl ? __builtin_inff() : fmaxf(acc[j], EPS_F);
            minv[j] = fminf(minv[j], d2);
        }

        __syncthreads();      // all waves done reading buf before it is restaged
    }

    // d2 >= EPS > 0 so float bits are order-preserving -> atomic u32 min
    #pragma unroll
    for (int j = 0; j < 8; ++j) {
        const int row = rowW + j + halfSel * 8;
        atomicMin(&dn2min[row], __float_as_uint(minv[j]));
    }
}

// ---------------------------------------------------------------------------
// Kernel 4: per-sample d_p, softplus, accumulate. One block per sample.
// ---------------------------------------------------------------------------
__global__ __launch_bounds__(256) void cl_loss_kernel(
    const float* __restrict__ feat, const int* __restrict__ labels,
    const float* __restrict__ sums, const float* __restrict__ counts,
    const unsigned int* __restrict__ dn2min, const int* __restrict__ epoch_p,
    float* __restrict__ accum)
{
    __shared__ float red[256];
    const int s = blockIdx.x;
    const int i = threadIdx.x;
    const float x = feat[(size_t)s * D_DIM + i];
    red[i] = x * x;
    __syncthreads();
    #pragma unroll
    for (int t = 128; t > 0; t >>= 1) {
        if (i < t) red[i] += red[i + t];
        __syncthreads();
    }
    const float nrm2 = red[0];
    __syncthreads();

    const float f   = x / fmaxf(sqrtf(nrm2), EPS_F);
    const int   lab = labels[s];
    const float cnt = counts[lab];
    const float cp  = (sums[(size_t)lab * D_DIM + i] - f) / fmaxf(cnt - 1.0f, 1.0f);
    const float dd  = f - cp;
    red[i] = dd * dd;
    __syncthreads();
    #pragma unroll
    for (int t = 128; t > 0; t >>= 1) {
        if (i < t) red[i] += red[i + t];
        __syncthreads();
    }

    if (i == 0) {
        const float dp    = sqrtf(red[0] + EPS_F);
        const float dn2   = __uint_as_float(dn2min[s]);
        const bool  fin   = (dn2 < __builtin_inff());
        const float dn    = fin ? sqrtf(dn2) : 0.0f;
        const bool  valid = (cnt > 1.0f) && fin;
        const float temp  = fminf(0.3f + 0.02f * (float)(*epoch_p), 1.0f);
        const float z     = (dp - dn) / temp;
        const float sp    = (z > 0.0f) ? (z + log1pf(expf(-z))) : log1pf(expf(z));
        if (valid) {
            atomicAdd(&accum[0], sp);
            atomicAdd(&accum[1], 1.0f);
        }
    }
}

// ---------------------------------------------------------------------------
// Kernel 5: finalize scalar loss
// ---------------------------------------------------------------------------
__global__ void cl_finalize_kernel(const float* __restrict__ accum,
                                   float* __restrict__ out)
{
    const float sum = accum[0];
    const float cnt = accum[1];
    out[0] = (cnt > 0.0f) ? (sum / fmaxf(cnt, 1.0f)) : 0.0f;   // WEIGHT = 1.0
}

// ---------------------------------------------------------------------------
extern "C" void kernel_launch(void* const* d_in, const int* in_sizes, int n_in,
                              void* d_out, int out_size, void* d_ws, size_t ws_size,
                              hipStream_t stream)
{
    const float* feat    = (const float*)d_in[0];
    const int*   labels  = (const int*)d_in[1];
    const int*   epoch_p = (const int*)d_in[2];
    float*       out     = (float*)d_out;

    char* ws = (char*)d_ws;
    _Float16*     fn16   = (_Float16*)(ws + OFF_FN16);
    _Float16*     cent16 = (_Float16*)(ws + OFF_CENT16);
    float*        sums   = (float*)(ws + OFF_SUMS);
    float*        counts = (float*)(ws + OFF_COUNTS);
    float*        cn2    = (float*)(ws + OFF_CN2);
    unsigned int* dn2min = (unsigned int*)(ws + OFF_DN2);
    float*        accum  = (float*)(ws + OFF_ACCUM);

    cl_init_kernel<<<dim3((C_DIM * D_DIM + 255) / 256), dim3(256), 0, stream>>>(
        sums, counts, dn2min, accum);

    cl_norm_scatter_kernel<<<dim3(B_DIM), dim3(256), 0, stream>>>(
        feat, labels, fn16, sums, counts);

    cl_centroid_kernel<<<dim3(C_DIM), dim3(256), 0, stream>>>(
        sums, counts, cent16, cn2);

    cl_gemm_min_kernel<<<dim3(B_DIM / 128), dim3(256), 0, stream>>>(
        fn16, cent16, cn2, counts, labels, dn2min);

    cl_loss_kernel<<<dim3(B_DIM), dim3(256), 0, stream>>>(
        feat, labels, sums, counts, dn2min, epoch_p, accum);

    cl_finalize_kernel<<<dim3(1), dim3(1), 0, stream>>>(accum, out);
}
